// SelfAttention_46119358825182
// MI455X (gfx1250) — compile-verified, matrix operand order checked by reading, not run on hardware
//
#include <hip/hip_runtime.h>
#include <hip/hip_bf16.h>

// ---------------------------------------------------------------------------
// Self-attention for MI455X (gfx1250, wave32, WMMA bf16 16x16x32).
// B=4, S=2048, D=512, H=8, DV=64.
// Bound by the mandatory 536 MB norms write (B,H,S,S fp32); all GEMMs run on
// v_wmma_f32_16x16x32_bf16 with fp32 accumulation. Softmax statistics are
// tracked per-lane (no shuffles in the streaming loops) and merged once.
// ---------------------------------------------------------------------------

#define BB 4
#define SS 2048
#define DD 512
#define HH 8
#define DVV 64

typedef __attribute__((ext_vector_type(16))) __bf16 v16bf;
typedef __attribute__((ext_vector_type(8)))  __bf16 v8bf;
typedef __attribute__((ext_vector_type(8)))  float  v8f;

// A-fragment (16x32 bf16, row-major source) per CDNA5 ISA 16-bit A layout:
//   lane L holds row M = L%16; lanes 0-15 K-octets {0..7,16..23},
//   lanes 16-31 K-octets {8..15,24..31}; VGPR i holds K pair (f[2i],f[2i+1]).
// B-fragments are loaded as the A-layout of B^T (so W, K-rows and V^T all
// use this same loader).
__device__ __forceinline__ v16bf load_frag(const __bf16* __restrict__ base, int ld) {
  const int lane = threadIdx.x & 31;
  const int row  = lane & 15;
  const int kb   = (lane & 16) ? 8 : 0;
  const __bf16* p = base + (size_t)row * ld + kb;
  v8bf lo = *(const v8bf*)(p);
  v8bf hi = *(const v8bf*)(p + 16);
  v16bf f;
#pragma unroll
  for (int i = 0; i < 8; ++i) { f[i] = lo[i]; f[i + 8] = hi[i]; }
  return f;
}

__device__ __forceinline__ v8f wmma_bf16(v16bf a, v16bf b, v8f c) {
  // (neg_a, A, neg_b, B, c_mod, C, reuse_a, reuse_b)
  return __builtin_amdgcn_wmma_f32_16x16x32_bf16(false, a, false, b,
                                                 (short)0, c, false, false);
}

// ---------------------------------------------------------------------------
__global__ void cvt_bf16_kernel(const float* __restrict__ in,
                                __bf16* __restrict__ out, int n) {
  int i = blockIdx.x * blockDim.x + threadIdx.x;
  for (; i < n; i += gridDim.x * blockDim.x) out[i] = (__bf16)in[i];
}

// ---------------------------------------------------------------------------
// y = X (MxK) @ W^T (W is NxK row-major, torch Linear) + bias.
// Wave tile = 16(M) x 64(N). Grid: (M/128, N/64), 256 threads (8 waves).
// mode 0: out bf16 (B,H,S,DV)   [Q,K]
// mode 2: out bf16 (B,H,DV,S)   [V, transposed for P.V B-fragments]
// mode 1: out fp32 flat M*N     [final projection -> d_out]
__global__ __launch_bounds__(256) void gemm_xwT_kernel(
    const __bf16* __restrict__ X, const __bf16* __restrict__ W,
    const float* __restrict__ bias, void* __restrict__ out,
    int M, int N, int K, int mode)
{
  const int wave  = threadIdx.x >> 5;
  const int lane  = threadIdx.x & 31;
  const int m0    = (blockIdx.x * 8 + wave) * 16;
  const int n0    = blockIdx.y * 64;
  const int col   = lane & 15;
  const int rbase = (lane & 16) ? 8 : 0;

  v8f acc[4];
#pragma unroll
  for (int t = 0; t < 4; ++t) {
    float bv = bias[n0 + t * 16 + col];
#pragma unroll
    for (int v = 0; v < 8; ++v) acc[t][v] = bv;
  }

#pragma unroll 2
  for (int k = 0; k < K; k += 32) {
    v16bf a = load_frag(X + (size_t)m0 * K + k, K);
#pragma unroll
    for (int t = 0; t < 4; ++t) {
      v16bf b = load_frag(W + (size_t)(n0 + t * 16) * K + k, K);
      acc[t] = wmma_bf16(a, b, acc[t]);
    }
  }

#pragma unroll
  for (int t = 0; t < 4; ++t) {
#pragma unroll
    for (int v = 0; v < 8; ++v) {
      const int m = m0 + rbase + v;
      const int n = n0 + t * 16 + col;
      const float val = acc[t][v];
      if (mode == 1) {
        ((float*)out)[(size_t)m * N + n] = val;
      } else {
        const int b = m >> 11, s = m & (SS - 1);
        const int h = n >> 6,  d = n & (DVV - 1);
        if (mode == 0)
          ((__bf16*)out)[((size_t)(b * HH + h) * SS + s) * DVV + d] = (__bf16)val;
        else
          ((__bf16*)out)[((size_t)(b * HH + h) * DVV + d) * SS + s] = (__bf16)val;
      }
    }
  }
}

// ---------------------------------------------------------------------------
__global__ void vnorm_kernel(const __bf16* __restrict__ Vt,
                             float* __restrict__ vnorm) {
  int idx = blockIdx.x * blockDim.x + threadIdx.x;   // over B*H*S
  if (idx >= BB * HH * SS) return;
  const int bh = idx >> 11;
  const int j  = idx & (SS - 1);
  const __bf16* p = Vt + (size_t)bh * DVV * SS + j;
  float s = 0.f;
#pragma unroll
  for (int d = 0; d < DVV; ++d) { float x = (float)p[(size_t)d * SS]; s += x * x; }
  vnorm[idx] = sqrtf(s);
}

// ---------------------------------------------------------------------------
// Fused attention: softmax(QK^T/8), norms = P * ||V_j||, v_bar = P.V.
// Each wave owns one 16-row i-tile; block = 8 waves = 128 rows of one (b,h).
// Grid: B*H * (S/128) = 512 blocks, 256 threads.
__global__ __launch_bounds__(256) void attn_kernel(
    const __bf16* __restrict__ Qh,    // (B*H, S, DV)
    const __bf16* __restrict__ Kh,    // (B*H, S, DV)
    const __bf16* __restrict__ Vt,    // (B*H, DV, S)
    const float*  __restrict__ vnorm, // (B*H, S)
    float* __restrict__ norms_out,    // (B*H, S, S)
    __bf16* __restrict__ vbar)        // (B, S, D) heads merged
{
  __shared__ __bf16 lds_p[8][16 * 32];   // per-wave 16x32 P tile (C->A reshape)

  const int wave  = threadIdx.x >> 5;
  const int lane  = threadIdx.x & 31;
  const int bh    = blockIdx.x >> 4;                 // S/128 = 16 i-blocks
  const int i0    = (blockIdx.x & 15) * 128 + wave * 16;
  const int col   = lane & 15;
  const int rbase = (lane & 16) ? 8 : 0;
  const float rscale = 0.125f;                       // 1/sqrt(DV)

  const __bf16* Qb = Qh + (size_t)bh * SS * DVV;
  const __bf16* Kb = Kh + (size_t)bh * SS * DVV;
  const __bf16* Vb = Vt + (size_t)bh * DVV * SS;
  const float*  vn = vnorm + (size_t)bh * SS;

  const v16bf a0 = load_frag(Qb + (size_t)i0 * DVV + 0,  DVV);
  const v16bf a1 = load_frag(Qb + (size_t)i0 * DVV + 32, DVV);

  // Per-lane online softmax stats: lane tracks max/sum over its own 128-column
  // subset of each of its 8 rows (C-layout: row = rbase+v, col = lane%16+16k).
  // No cross-lane traffic in the streaming loop.
  float mrow[8], lrow[8];
#pragma unroll
  for (int v = 0; v < 8; ++v) { mrow[v] = -1e30f; lrow[v] = 0.f; }

  // ---- pass 1: online per-lane max/sum (E never touches HBM) ----
#pragma unroll 2
  for (int j0 = 0; j0 < SS; j0 += 16) {
    __builtin_prefetch(Kb + (size_t)(j0 + 64) * DVV, 0, 0);
    v16bf b0 = load_frag(Kb + (size_t)j0 * DVV + 0,  DVV);
    v16bf b1 = load_frag(Kb + (size_t)j0 * DVV + 32, DVV);
    v8f c = {};
    c = wmma_bf16(a0, b0, c);
    c = wmma_bf16(a1, b1, c);
#pragma unroll
    for (int v = 0; v < 8; ++v) {
      float e = c[v] * rscale;
      float mnew = fmaxf(mrow[v], e);
      lrow[v] = lrow[v] * __expf(mrow[v] - mnew) + __expf(e - mnew);
      mrow[v] = mnew;
    }
  }

  // ---- one cross-lane (m,l) log-sum-exp merge over the 16-lane row group ----
#pragma unroll
  for (int v = 0; v < 8; ++v) {
#pragma unroll
    for (int s = 1; s < 16; s <<= 1) {
      float om = __shfl_xor(mrow[v], s, 32);
      float ol = __shfl_xor(lrow[v], s, 32);
      float mn = fmaxf(mrow[v], om);
      lrow[v] = lrow[v] * __expf(mrow[v] - mn) + ol * __expf(om - mn);
      mrow[v] = mn;
    }
  }
  float linv[8];
#pragma unroll
  for (int v = 0; v < 8; ++v) linv[v] = 1.f / lrow[v];

  v8f O[4];
#pragma unroll
  for (int t = 0; t < 4; ++t)
#pragma unroll
    for (int v = 0; v < 8; ++v) O[t][v] = 0.f;

  // ---- pass 2: recompute E, emit norms, accumulate v_bar = P.V ----
  for (int jt = 0; jt < SS; jt += 32) {
    __builtin_prefetch(Vb + (size_t)(jt + 32), 0, 0);
#pragma unroll
    for (int half = 0; half < 2; ++half) {
      const int j0 = jt + half * 16;
      v16bf b0 = load_frag(Kb + (size_t)j0 * DVV + 0,  DVV);
      v16bf b1 = load_frag(Kb + (size_t)j0 * DVV + 32, DVV);
      v8f c = {};
      c = wmma_bf16(a0, b0, c);
      c = wmma_bf16(a1, b1, c);
      const float vnj = vn[j0 + col];
#pragma unroll
      for (int v = 0; v < 8; ++v) {
        float p = __expf(c[v] * rscale - mrow[v]) * linv[v];
        const int i = i0 + rbase + v;
        norms_out[((size_t)bh * SS + i) * SS + (j0 + col)] = p * vnj;
        lds_p[wave][(rbase + v) * 32 + half * 16 + col] = (__bf16)p;
      }
    }
    __syncthreads();
    v16bf pf = load_frag(&lds_p[wave][0], 32);
#pragma unroll
    for (int dt = 0; dt < 4; ++dt) {
      v16bf bv = load_frag(Vb + (size_t)(dt * 16) * SS + jt, SS);
      O[dt] = wmma_bf16(pf, bv, O[dt]);
    }
    __syncthreads();
  }

  // ---- epilogue: v_bar -> (B, S, H*DV) bf16 ----
  const int b = bh >> 3, h = bh & 7;
#pragma unroll
  for (int dt = 0; dt < 4; ++dt)
#pragma unroll
    for (int v = 0; v < 8; ++v) {
      const int i = i0 + rbase + v;
      const int d = h * DVV + dt * 16 + col;
      vbar[(size_t)(b * SS + i) * DD + d] = (__bf16)O[dt][v];
    }
}

// ---------------------------------------------------------------------------
extern "C" void kernel_launch(void* const* d_in, const int* in_sizes, int n_in,
                              void* d_out, int out_size, void* d_ws, size_t ws_size,
                              hipStream_t stream) {
  (void)in_sizes; (void)n_in; (void)out_size;
  const float* q   = (const float*)d_in[0];
  const float* kin = (const float*)d_in[1];
  const float* val = (const float*)d_in[2];
  const float* Wq  = (const float*)d_in[3]; const float* bq = (const float*)d_in[4];
  const float* Wk  = (const float*)d_in[5]; const float* bk = (const float*)d_in[6];
  const float* Wv  = (const float*)d_in[7]; const float* bv = (const float*)d_in[8];
  const float* Wo  = (const float*)d_in[9]; const float* bo = (const float*)d_in[10];

  float* x_out     = (float*)d_out;                       // (B,S,D)
  float* norms_out = x_out + (size_t)BB * SS * DD;        // (B,H,S,S)

  const int M = BB * SS, N = DD, K = DD;                  // 8192 x 512 x 512
  const int NX = M * K;                                   // 4,194,304
  const int NW = DD * DD;                                 // 262,144

  // workspace carve-up (~58.5 MB total)
  char* ws = (char*)d_ws; size_t off = 0;
  auto wsa = [&](size_t bytes) { void* p = ws + off; off += (bytes + 255) & ~(size_t)255; return p; };
  __bf16* qbf  = (__bf16*)wsa((size_t)NX * 2);
  __bf16* kbf  = (__bf16*)wsa((size_t)NX * 2);
  __bf16* vbf  = (__bf16*)wsa((size_t)NX * 2);
  __bf16* wqb  = (__bf16*)wsa((size_t)NW * 2);
  __bf16* wkb  = (__bf16*)wsa((size_t)NW * 2);
  __bf16* wvb  = (__bf16*)wsa((size_t)NW * 2);
  __bf16* wob  = (__bf16*)wsa((size_t)NW * 2);
  __bf16* Qh   = (__bf16*)wsa((size_t)NX * 2);            // (B,H,S,DV)
  __bf16* Kh   = (__bf16*)wsa((size_t)NX * 2);            // (B,H,S,DV)
  __bf16* Vt   = (__bf16*)wsa((size_t)NX * 2);            // (B,H,DV,S)
  __bf16* vbar = (__bf16*)wsa((size_t)NX * 2);            // (B,S,D)
  float*  vnrm = (float*) wsa((size_t)BB * HH * SS * 4);  // (B,H,S)
  (void)ws_size;

  // 1) fp32 -> bf16
  cvt_bf16_kernel<<<1024, 256, 0, stream>>>(q,   qbf, NX);
  cvt_bf16_kernel<<<1024, 256, 0, stream>>>(kin, kbf, NX);
  cvt_bf16_kernel<<<1024, 256, 0, stream>>>(val, vbf, NX);
  cvt_bf16_kernel<<<256,  256, 0, stream>>>(Wq, wqb, NW);
  cvt_bf16_kernel<<<256,  256, 0, stream>>>(Wk, wkb, NW);
  cvt_bf16_kernel<<<256,  256, 0, stream>>>(Wv, wvb, NW);
  cvt_bf16_kernel<<<256,  256, 0, stream>>>(Wo, wob, NW);

  // 2) projections (WMMA bf16)
  dim3 ggrid(M / 128, N / 64);
  gemm_xwT_kernel<<<ggrid, 256, 0, stream>>>(qbf, wqb, bq, Qh, M, N, K, 0);
  gemm_xwT_kernel<<<ggrid, 256, 0, stream>>>(kbf, wkb, bk, Kh, M, N, K, 0);
  gemm_xwT_kernel<<<ggrid, 256, 0, stream>>>(vbf, wvb, bv, Vt, M, N, K, 2);

  // 3) per-row V norms
  vnorm_kernel<<<(BB * HH * SS) / 256, 256, 0, stream>>>(Vt, vnrm);

  // 4) fused attention (softmax + norms + P.V)
  attn_kernel<<<BB * HH * (SS / 128), 256, 0, stream>>>(Qh, Kh, Vt, vnrm,
                                                        norms_out, vbar);

  // 5) output projection -> fp32 x
  gemm_xwT_kernel<<<ggrid, 256, 0, stream>>>(vbar, wob, bo, x_out, M, N, K, 1);
}
